// HelgasonLosslessLayer_14654428414608
// MI455X (gfx1250) — matrix operand order repaired and checked by reading, not compiled, not run on hardware
//
#include <hip/hip_runtime.h>
#include <math.h>

// ---------------------------------------------------------------------------
// Types for CDNA5 WMMA (wave32)
// ---------------------------------------------------------------------------
typedef __attribute__((ext_vector_type(16))) __bf16 v16bf;
typedef __attribute__((ext_vector_type(8)))  float  v8f;

#define MAX_L      8
#define NLM        81          // (MAX_L+1)^2
#define R_CONST    0.95f
#define ENERGY_THR 0.9999f
#define GOLDEN_F   0.618033988749f
#define PI_F       3.14159265358979f

// --------- CDNA5 async global->LDS support (guarded; sync fallback) --------
#if defined(__has_builtin)
#  if __has_builtin(__builtin_amdgcn_global_load_async_to_lds_b128)
#    define HAVE_ASYNC_LDS 1
#  endif
#endif
#ifndef HAVE_ASYNC_LDS
#  define HAVE_ASYNC_LDS 0
#endif

#if defined(__has_builtin)
#  if __has_builtin(__builtin_amdgcn_s_wait_asynccnt)
#    define WAIT_ASYNCCNT(n) __builtin_amdgcn_s_wait_asynccnt(n)
#  else
#    define WAIT_ASYNCCNT(n) asm volatile("s_wait_asynccnt %0" ::"n"(n))
#  endif
#else
#  define WAIT_ASYNCCNT(n) asm volatile("s_wait_asynccnt %0" ::"n"(n))
#endif

// builtin expects pointers to GCC-style int4 vectors in AS1 (global) / AS3 (LDS)
typedef int async_v4i __attribute__((vector_size(16)));
#define AS_GLOBAL_V4(p) ((__attribute__((address_space(1))) async_v4i*)(p))
#define AS_LOCAL_V4(p)  ((__attribute__((address_space(3))) async_v4i*)(p))

// f32 -> bf16 round-to-nearest-even
static __device__ __forceinline__ unsigned short f2bf(float f) {
    unsigned u = __float_as_uint(f);
    unsigned r = 0x7FFFu + ((u >> 16) & 1u);
    u += r;
    return (unsigned short)(u >> 16);
}

// idx in the flattened LM list -> (l, m), idx in [l^2, (l+1)^2)
static __device__ __forceinline__ void idx2lm(int idx, int* l, int* m) {
    int ll = 0;
    while ((ll + 1) * (ll + 1) <= idx) ll++;
    *l = ll;
    *m = idx - ll * ll - ll;
}

// norm = sqrt((2l+1) * (l-|m|)!/(l+|m|)! / (4*pi))
static __device__ float ylm_norm(int l, int am) {
    float ratio = 1.f;
    for (int k = l - am + 1; k <= l + am; ++k) ratio /= (float)k;
    return sqrtf((float)(2 * l + 1) * ratio / (4.f * PI_F));
}

// Associated Legendre table P[l][m] for all 0<=m<=l<=MAX_L (reference recurrences)
static __device__ void legendre_table(float x, float P[MAX_L + 1][MAX_L + 1]) {
    float somx2 = sqrtf(fmaxf(0.f, 1.f - x * x));
    for (int m = 0; m <= MAX_L; ++m) {
        float pmm = 1.f, fact = 1.f;
        for (int i = 0; i < m; ++i) { pmm *= -fact * somx2; fact += 2.f; }
        P[m][m] = pmm;
        if (m < MAX_L) {
            float pmmp1 = x * (float)(2 * m + 1) * pmm;
            P[m + 1][m] = pmmp1;
            float pll = pmm, plp1 = pmmp1;
            for (int l = m + 2; l <= MAX_L; ++l) {
                float pnew = ((float)(2 * l - 1) * x * plp1 - (float)(l + m - 1) * pll) / (float)(l - m);
                P[l][m] = pnew; pll = plp1; plp1 = pnew;
            }
        }
    }
}

static __device__ __forceinline__ void sphere_coords(float zr, float zi,
                                                     float* cos_t, float* phi) {
    float r = sqrtf(zr * zr + zi * zi);
    *phi = atan2f(zi, zr);
    float rm = fminf(r, 0.99f);
    *cos_t = cosf(2.f * atanhf(rm) * PI_F / (2.f * (float)MAX_L));
}

// ---------------------------------------------------------------------------
// 0) zero small scratch
// ---------------------------------------------------------------------------
__global__ void zero_kernel(float* __restrict__ p, int n) {
    int i = blockIdx.x * blockDim.x + threadIdx.x;
    int s = gridDim.x * blockDim.x;
    for (; i < n; i += s) p[i] = 0.f;
}

// ---------------------------------------------------------------------------
// 1) fuse: dst = imp0*s0 + imp1*s1, imp = softmax([0, 0.5])
// ---------------------------------------------------------------------------
__global__ void fuse_kernel(const float* __restrict__ s0, const float* __restrict__ s1,
                            float* __restrict__ dst, int n) {
    float e1 = expf(0.5f);
    float i1 = e1 / (1.f + e1);
    float i0 = 1.f / (1.f + e1);
    int i = blockIdx.x * blockDim.x + threadIdx.x;
    int s = gridDim.x * blockDim.x;
    for (; i < n; i += s) dst[i] = i0 * s0[i] + i1 * s1[i];
}

// ---------------------------------------------------------------------------
// 2) scale = max over pairs of sqrt(re^2+im^2)
// ---------------------------------------------------------------------------
__global__ void scale_kernel(const float* __restrict__ flat, int np,
                             unsigned* __restrict__ scale_bits) {
    float mx = 0.f;
    int i = blockIdx.x * blockDim.x + threadIdx.x;
    int s = gridDim.x * blockDim.x;
    for (; i < np; i += s) {
        float re = flat[2 * i], im = flat[2 * i + 1];
        mx = fmaxf(mx, sqrtf(re * re + im * im));
    }
    for (int off = 16; off > 0; off >>= 1)
        mx = fmaxf(mx, __shfl_down(mx, off, 32));
    if ((threadIdx.x & 31) == 0) atomicMax(scale_bits, __float_as_uint(mx));
}

// ---------------------------------------------------------------------------
// 3) forward transform: accumulate per-(l,m) sums of Y_lm over all points
// ---------------------------------------------------------------------------
__global__ void fwd_coeff_kernel(const float* __restrict__ flat, int np,
                                 const unsigned* __restrict__ scale_bits,
                                 float* __restrict__ coef_sums) {
    __shared__ float s_norm[NLM];
    if (threadIdx.x < NLM) {
        int l, m; idx2lm(threadIdx.x, &l, &m);
        s_norm[threadIdx.x] = ylm_norm(l, m < 0 ? -m : m);
    }
    __syncthreads();

    float scale = __uint_as_float(*scale_bits);
    float safe  = scale > 0.f ? scale : 1.f;

    float acc[NLM];
    for (int i = 0; i < NLM; ++i) acc[i] = 0.f;

    int i = blockIdx.x * blockDim.x + threadIdx.x;
    int s = gridDim.x * blockDim.x;
    for (; i < np; i += s) {
        float re = flat[2 * i], im = flat[2 * i + 1];
        float a = re / safe, b = im / safe;
        float den = coshf(2.f * a) + cosf(2.f * b);
        float pr  = scale > 0.f ? sinhf(2.f * a) / den * R_CONST : re;
        float pim = scale > 0.f ? sinf(2.f * b) / den * R_CONST : im;
        float ct, phi; sphere_coords(pr, pim, &ct, &phi);

        float P[MAX_L + 1][MAX_L + 1];
        legendre_table(ct, P);
        float cm[MAX_L + 1], sm[MAX_L + 1];
        cm[0] = 1.f; sm[0] = 0.f;
        float c1 = cosf(phi), s1 = sinf(phi);
        for (int m = 1; m <= MAX_L; ++m) {
            cm[m] = cm[m - 1] * c1 - sm[m - 1] * s1;
            sm[m] = sm[m - 1] * c1 + cm[m - 1] * s1;
        }
        int idx = 0;
        for (int l = 0; l <= MAX_L; ++l)
            for (int mm = -l; mm <= l; ++mm, ++idx) {
                int am = mm < 0 ? -mm : mm;
                float ang = mm >= 0 ? cm[am] : sm[am];
                acc[idx] += s_norm[idx] * P[l][am] * ang;
            }
    }

    int lane = threadIdx.x & 31;
    for (int idx = 0; idx < NLM; ++idx) {
        float v = acc[idx];
        for (int off = 16; off > 0; off >>= 1) v += __shfl_down(v, off, 32);
        if (lane == 0) atomicAdd(&coef_sums[idx], v);
    }
}

// ---------------------------------------------------------------------------
// 4) compress (81 coefficients, single thread)
// ---------------------------------------------------------------------------
__global__ void compress_kernel(float* __restrict__ coef, int np) {
    if (threadIdx.x != 0 || blockIdx.x != 0) return;
    float c[NLM], mag[NLM];
    for (int i = 0; i < NLM; ++i) { c[i] = coef[i] / (float)np; mag[i] = fabsf(c[i]); }
    int rank[NLM];
    for (int i = 0; i < NLM; ++i) {
        int r = 0;
        for (int j = 0; j < NLM; ++j)
            if (mag[j] > mag[i] || (mag[j] == mag[i] && j < i)) r++;
        rank[i] = r;
    }
    float sm2[NLM];
    for (int i = 0; i < NLM; ++i) sm2[rank[i]] = mag[i] * mag[i];
    float tot = 0.f;
    for (int p = 0; p < NLM; ++p) tot += sm2[p];
    int kc;
    if (tot > 0.f) {
        float mx = fmaxf(tot, 1e-30f);
        float cum = 0.f; int cnt = 0;
        for (int p = 0; p < NLM; ++p) { cum += sm2[p]; if (cum / mx <= ENERGY_THR) cnt++; }
        kc = cnt + 1;
    } else kc = NLM;
    if (kc < NLM / 2) kc = NLM / 2;
    for (int i = 0; i < NLM; ++i) coef[i] = (rank[i] < kc) ? c[i] : 0.f;
}

// ---------------------------------------------------------------------------
// 5) inverse transform -> PACKED bf16 weights
// ---------------------------------------------------------------------------
static __device__ float recon_point(int i, int np, const float* s_c,
                                    const float* s_norm, float scale) {
    float fi = (float)i;
    float r = fminf((fi + 0.5f) / (float)np * R_CONST, 0.95f);
    float theta = fmodf(2.f * PI_F * fi * GOLDEN_F, 1.f);
    float zr = r * cosf(theta), zi = r * sinf(theta);
    float ct, phi; sphere_coords(zr, zi, &ct, &phi);

    float P[MAX_L + 1][MAX_L + 1];
    legendre_table(ct, P);
    float cm[MAX_L + 1], sm[MAX_L + 1];
    cm[0] = 1.f; sm[0] = 0.f;
    float c1 = cosf(phi), s1 = sinf(phi);
    for (int m = 1; m <= MAX_L; ++m) {
        cm[m] = cm[m - 1] * c1 - sm[m - 1] * s1;
        sm[m] = sm[m - 1] * c1 + cm[m - 1] * s1;
    }
    float recon = 0.f;
    int idx = 0;
    for (int l = 0; l <= MAX_L; ++l)
        for (int mm = -l; mm <= l; ++mm, ++idx) {
            int am = mm < 0 ? -mm : mm;
            float ang = mm >= 0 ? cm[am] : sm[am];
            recon += s_c[idx] * s_norm[idx] * P[l][am] * ang;
        }
    return scale > 0.f ? recon * scale : recon;
}

// fc_mode==1: original (3072x768), keep even d (384 packed cols) AND even f rows
//             -> output (1536 x 384); only np/2 points evaluated.
// fc_mode==0: original (768x3072), keep even f (1536 packed cols)
//             -> output (768 x 1536) = all np points in order.
__global__ void inv_kernel(const float* __restrict__ coef,
                           const unsigned* __restrict__ scale_bits,
                           int np, unsigned short* __restrict__ Wout, int fc_mode) {
    __shared__ float s_norm[NLM];
    __shared__ float s_c[NLM];
    if (threadIdx.x < NLM) {
        int l, m; idx2lm(threadIdx.x, &l, &m);
        s_norm[threadIdx.x] = ylm_norm(l, m < 0 ? -m : m);
        float cv = coef[threadIdx.x];
        s_c[threadIdx.x] = fabsf(cv) > 1e-10f ? cv : 0.f;
    }
    __syncthreads();

    float scale = __uint_as_float(*scale_bits);
    int t = blockIdx.x * blockDim.x + threadIdx.x;
    int s = gridDim.x * blockDim.x;
    if (fc_mode) {
        int nq = np / 2;                       // 1536*384 outputs
        for (int q = t; q < nq; q += s) {
            int i = (q / 384) * 768 + (q % 384);   // even-row, even-col point index
            Wout[q] = f2bf(recon_point(i, np, s_c, s_norm, scale));
        }
    } else {
        for (int i = t; i < np; i += s)        // 768*1536 outputs
            Wout[i] = f2bf(recon_point(i, np, s_c, s_norm, scale));
    }
}

// ---------------------------------------------------------------------------
// 6a) pack even-d columns of x -> bf16 (8192 x 384)
// ---------------------------------------------------------------------------
__global__ void pack_x_kernel(const float* __restrict__ x,
                              unsigned short* __restrict__ xp, int total) {
    int i = blockIdx.x * blockDim.x + threadIdx.x;
    int s = gridDim.x * blockDim.x;
    for (; i < total; i += s) {
        int row = i / 384, col = i % 384;
        xp[i] = f2bf(x[(size_t)row * 768 + 2 * col]);
    }
}

// 6b) pack even entries of a vector: out[q] = in[2q]
__global__ void pack_even_kernel(const float* __restrict__ in,
                                 float* __restrict__ out, int n) {
    int i = blockIdx.x * blockDim.x + threadIdx.x;
    int s = gridDim.x * blockDim.x;
    for (; i < n; i += s) out[i] = in[2 * i];
}

// ---------------------------------------------------------------------------
// 7) WMMA GEMM:  C(MxN) = A(MxK) * B(NxK)^T, bf16 in, f32 accum.
//    Workgroup tile 128x128, 8 waves (4 in M x 2 in N), wave tile 32x64.
//    Double-buffered LDS; async global->LDS when available.
//    EPI==0: C = bf16( gelu_exact(acc + bias[col]) )
//    EPI==1: C = f32 ( acc + bias[col] )
// ---------------------------------------------------------------------------
template <int EPI>
__global__ __launch_bounds__(256)
void gemm_bf16_wmma(const unsigned short* __restrict__ A,
                    const unsigned short* __restrict__ B,
                    const float* __restrict__ bias,
                    void* __restrict__ Cout,
                    int M, int N, int K) {
    __shared__ __align__(16) unsigned short As[2][128 * 32];
    __shared__ __align__(16) unsigned short Bs[2][128 * 32];

    const int t    = threadIdx.x;
    const int wave = t >> 5;
    const int lane = t & 31;
    const int half = lane >> 4;
    const int l16  = lane & 15;
    const int waveM = (wave & 3) * 32;   // 4 waves tile M=128
    const int waveN = (wave >> 2) * 64;  // 2 waves tile N=128
    const int blockM = blockIdx.y * 128;
    const int blockN = blockIdx.x * 128;

    v8f acc[2][4];
    for (int a = 0; a < 2; ++a)
        for (int b = 0; b < 4; ++b)
            for (int e = 0; e < 8; ++e) acc[a][b][e] = 0.f;

    // cooperative tile loader: thread t moves 16 bf16 (2x b128) of one row
    const int ldRow = t >> 1;
    const int ldCol = (t & 1) << 4;
    const unsigned short* gA = A + (size_t)(blockM + ldRow) * K + ldCol;
    const unsigned short* gB = B + (size_t)(blockN + ldRow) * K + ldCol;
    const int lOff = ldRow * 32 + ldCol;

    auto issueTile = [&](int buf, int k0) {
#if HAVE_ASYNC_LDS
        __builtin_amdgcn_global_load_async_to_lds_b128(
            AS_GLOBAL_V4(gA + k0),     AS_LOCAL_V4(&As[buf][lOff]),     0, 0);
        __builtin_amdgcn_global_load_async_to_lds_b128(
            AS_GLOBAL_V4(gA + k0 + 8), AS_LOCAL_V4(&As[buf][lOff + 8]), 0, 0);
        __builtin_amdgcn_global_load_async_to_lds_b128(
            AS_GLOBAL_V4(gB + k0),     AS_LOCAL_V4(&Bs[buf][lOff]),     0, 0);
        __builtin_amdgcn_global_load_async_to_lds_b128(
            AS_GLOBAL_V4(gB + k0 + 8), AS_LOCAL_V4(&Bs[buf][lOff + 8]), 0, 0);
#else
        const uint4* ga = reinterpret_cast<const uint4*>(gA + k0);
        uint4* sa = reinterpret_cast<uint4*>(&As[buf][lOff]);
        sa[0] = ga[0]; sa[1] = ga[1];
        const uint4* gb = reinterpret_cast<const uint4*>(gB + k0);
        uint4* sb = reinterpret_cast<uint4*>(&Bs[buf][lOff]);
        sb[0] = gb[0]; sb[1] = gb[1];
        __builtin_prefetch(gA + k0 + 32, 0, 1);
        __builtin_prefetch(gB + k0 + 32, 0, 1);
#endif
    };

    union Frag { v16bf v; unsigned u[8]; };

    const int nk = K / 32;
    issueTile(0, 0);
    for (int kt = 0; kt < nk; ++kt) {
        const int cur = kt & 1;
        if (kt + 1 < nk) issueTile(cur ^ 1, (kt + 1) * 32);
#if HAVE_ASYNC_LDS
        if (kt + 1 < nk) { WAIT_ASYNCCNT(4); } else { WAIT_ASYNCCNT(0); }
#endif
        __syncthreads();   // buf[cur] fully resident for all waves

        // B fragments: 32x16 bf16, lane holds col n=l16; K = (v<4 ? 2v : 16+2(v-4)) + slot + 8*half
        Frag bf[4];
        for (int tn = 0; tn < 4; ++tn) {
            int nrow = waveN + tn * 16 + l16;
            for (int v = 0; v < 8; ++v) {
                int kk = ((v < 4) ? 2 * v : 16 + 2 * (v - 4)) + 8 * half;
                bf[tn].u[v] = *reinterpret_cast<const unsigned*>(&Bs[cur][nrow * 32 + kk]);
            }
        }
        // A fragments: 16x32 bf16, lane holds row m=l16 (ISA 16-bit A layout)
        for (int tm = 0; tm < 2; ++tm) {
            Frag af;
            int mrow = waveM + tm * 16 + l16;
            for (int v = 0; v < 8; ++v) {
                int kk = ((v < 4) ? 2 * v : 16 + 2 * (v - 4)) + 8 * half;
                af.u[v] = *reinterpret_cast<const unsigned*>(&As[cur][mrow * 32 + kk]);
            }
            for (int tn = 0; tn < 4; ++tn) {
                acc[tm][tn] = __builtin_amdgcn_wmma_f32_16x16x32_bf16(
                    false, af.v, false, bf[tn].v, (short)0, acc[tm][tn], false, false);
            }
        }
        __syncthreads();   // protect buf[cur] before it is refilled at kt+2
    }

    // epilogue: C/D layout -> row = v + 8*half, col = l16
    for (int tm = 0; tm < 2; ++tm)
        for (int tn = 0; tn < 4; ++tn) {
            int col = blockN + waveN + tn * 16 + l16;
            float bv = bias[col];
            for (int v = 0; v < 8; ++v) {
                int row = blockM + waveM + tm * 16 + v + 8 * half;
                float val = acc[tm][tn][v] + bv;
                if (EPI == 0) {
                    val = 0.5f * val * (1.f + erff(val * 0.70710678118654752f));
                    reinterpret_cast<unsigned short*>(Cout)[(size_t)row * N + col] = f2bf(val);
                } else {
                    reinterpret_cast<float*>(Cout)[(size_t)row * N + col] = val;
                }
            }
        }
}

// ---------------------------------------------------------------------------
// host launcher
// ---------------------------------------------------------------------------
extern "C" void kernel_launch(void* const* d_in, const int* in_sizes, int n_in,
                              void* d_out, int out_size, void* d_ws, size_t ws_size,
                              hipStream_t stream) {
    const float* x      = (const float*)d_in[0];  // (8,1024,768) -> (8192,768)
    const float* w_fc   = (const float*)d_in[1];  // (2,3072,768)
    const float* w_proj = (const float*)d_in[2];  // (2,768,3072)
    const float* b_fc   = (const float*)d_in[3];  // (2,3072)
    const float* b_proj = (const float*)d_in[4];  // (2,768)

    const int DM = 768, DF = 3072, MSEQ = 8 * 1024;
    const int NW = DM * DF;       // 2359296
    const int NP = NW / 2;        // 1179648

    char* ws = (char*)d_ws;
    const size_t O_FUSED_FC = 0;                                   // f32 9437184
    const size_t O_FUSED_PR = O_FUSED_FC + (size_t)NW * 4;         // f32 9437184
    const size_t O_BIASFC   = O_FUSED_PR + (size_t)NW * 4;         // f32 12288
    const size_t O_BIASPR   = O_BIASFC + 12288;                    // f32 3072
    const size_t O_BIASFCE  = O_BIASPR + 3072;                     // f32 6144 (even f)
    const size_t O_COEF_FC  = O_BIASFCE + 6144;                    // 512
    const size_t O_COEF_PR  = O_COEF_FC + 512;                     // 512
    const size_t O_SCALE    = O_COEF_PR + 512;                     // 256
    const size_t O_WFC      = O_SCALE + 256;                       // bf16 1536x384 = 1179648
    const size_t O_WPR      = O_WFC + 1179648;                     // bf16 768x1536 = 2359296
    const size_t O_H        = O_WPR + 2359296;                     // bf16 8192x1536 = 25165824
    const size_t O_XP       = 0;   // bf16 8192x384 = 6291456, aliases dead fused buffers

    float*          fused_fc = (float*)(ws + O_FUSED_FC);
    float*          fused_pr = (float*)(ws + O_FUSED_PR);
    float*          bias_fc  = (float*)(ws + O_BIASFC);
    float*          bias_pr  = (float*)(ws + O_BIASPR);
    float*          bias_fce = (float*)(ws + O_BIASFCE);
    float*          coef_fc  = (float*)(ws + O_COEF_FC);
    float*          coef_pr  = (float*)(ws + O_COEF_PR);
    unsigned*       scale_fc = (unsigned*)(ws + O_SCALE);
    unsigned*       scale_pr = scale_fc + 1;
    unsigned short* Wfc      = (unsigned short*)(ws + O_WFC);   // packed (1536,384)
    unsigned short* Wpr      = (unsigned short*)(ws + O_WPR);   // packed (768,1536)
    unsigned short* hbuf     = (unsigned short*)(ws + O_H);     // (8192,1536)
    unsigned short* xp       = (unsigned short*)(ws + O_XP);    // (8192,384)

    // 0) zero coeff sums + scales
    zero_kernel<<<1, 256, 0, stream>>>((float*)(ws + O_COEF_FC), (512 + 512 + 256) / 4);

    // 1) fuse weights and biases
    fuse_kernel<<<1024, 256, 0, stream>>>(w_fc,   w_fc + NW,   fused_fc, NW);
    fuse_kernel<<<1024, 256, 0, stream>>>(w_proj, w_proj + NW, fused_pr, NW);
    fuse_kernel<<<12,   256, 0, stream>>>(b_fc,   b_fc + DF,   bias_fc,  DF);
    fuse_kernel<<<3,    256, 0, stream>>>(b_proj, b_proj + DM, bias_pr,  DM);
    pack_even_kernel<<<6, 256, 0, stream>>>(bias_fc, bias_fce, DF / 2);

    // 2) scales
    scale_kernel<<<512, 256, 0, stream>>>(fused_fc, NP, scale_fc);
    scale_kernel<<<512, 256, 0, stream>>>(fused_pr, NP, scale_pr);

    // 3) forward harmonic projection
    fwd_coeff_kernel<<<256, 256, 0, stream>>>(fused_fc, NP, scale_fc, coef_fc);
    fwd_coeff_kernel<<<256, 256, 0, stream>>>(fused_pr, NP, scale_pr, coef_pr);

    // 4) compress
    compress_kernel<<<1, 1, 0, stream>>>(coef_fc, NP);
    compress_kernel<<<1, 1, 0, stream>>>(coef_pr, NP);

    // 5) inverse transform -> packed bf16 weights
    inv_kernel<<<1024, 256, 0, stream>>>(coef_fc, scale_fc, NP, Wfc, 1);
    inv_kernel<<<1024, 256, 0, stream>>>(coef_pr, scale_pr, NP, Wpr, 0);

    // 6) pack x even-d columns -> bf16 (fused f32 buffers are dead; aliased)
    pack_x_kernel<<<2048, 256, 0, stream>>>(x, xp, MSEQ * (DM / 2));

    // 7) GEMM1: h_e = gelu(x_e @ Wfc_p^T + bias_fc_even)   M=8192 N=1536 K=384
    gemm_bf16_wmma<0><<<dim3(1536 / 128, MSEQ / 128), 256, 0, stream>>>(
        xp, Wfc, bias_fce, (void*)hbuf, MSEQ, 1536, 384);

    // 8) GEMM2: out = h_e @ Wpr_p^T + bias_proj             M=8192 N=768 K=1536
    gemm_bf16_wmma<1><<<dim3(DM / 128, MSEQ / 128), 256, 0, stream>>>(
        hbuf, Wpr, bias_pr, d_out, MSEQ, DM, 1536);
}
